// MoELayer_83708912599436
// MI455X (gfx1250) — matrix operand order repaired and checked by reading, not compiled, not run on hardware
//
#include <hip/hip_runtime.h>
#include <hip/hip_bf16.h>

// ---------------------------------------------------------------------------
// Switch-MoE (top-1) for MI455X / gfx1250, wave32 + WMMA bf16 + TDM gather.
//
// Shapes: B=2, S=2048 -> N=4096 tokens, D=768, F=3072, E=8.
// Routed FLOPs ~39 GFLOP. Weights (151 MB f32) are L2-resident (192 MB L2),
// so the limiter is L2->WGP weight traffic: with a 32-token tile each B
// fragment is reused by 2 WMMA tiles, halving weight fetch vs a 16-token
// tile (~2.4 GB total). The [N,F] ReLU intermediate never leaves LDS.
// Activation tiles (scattered token rows) are fetched by the Tensor Data
// Mover in gather mode (16 x 16-bit row indices per descriptor) with LDS
// row padding for bank-conflict-free b128 fragment reads.
// ---------------------------------------------------------------------------

constexpr int kB = 2, kS = 2048, kN = kB * kS;   // tokens
constexpr int kD = 768;                          // model dim
constexpr int kF = 3072;                         // ffn dim
constexpr int kE = 8;                            // experts
constexpr int kTileM  = 32;                      // tokens per workgroup tile
constexpr int kFChunk = 128;                     // F columns per iteration
constexpr int kXPitch = 780;                     // 768 + 3*4 dword TDM pad

typedef __attribute__((ext_vector_type(16))) __bf16 v16bf;
typedef __attribute__((ext_vector_type(8)))  float  v8f;
typedef __attribute__((ext_vector_type(4)))  unsigned int u32x4;
typedef __attribute__((ext_vector_type(8)))  int i32x8;
typedef __attribute__((ext_vector_type(4)))  int i32x4;

union ABFrag { v16bf v; uint4 q[2]; };           // 16 bf16 = 32B = 2 x b128

__device__ __forceinline__ __bf16 f2bf(float f) { return (__bf16)f; }

// Toolchain-portable TDM issue (5-arg on ROCm7.2/clang-22, 6-arg on clang-23).
#if defined(__clang_major__) && (__clang_major__ >= 23)
#define TDM_LOAD(g0, g1, g2, g3) \
    __builtin_amdgcn_tensor_load_to_lds((g0), (g1), (g2), (g3), \
                                        (i32x8){0,0,0,0,0,0,0,0}, 0)
#else
#define TDM_LOAD(g0, g1, g2, g3) \
    __builtin_amdgcn_tensor_load_to_lds((g0), (g1), (g2), (g3), 0)
#endif

// ---------------------------------------------------------------------------
// Kernel 1: router. One wave per token: logits = x . Wr, softmax, argmax.
// ---------------------------------------------------------------------------
__global__ void router_kernel(const float* __restrict__ x,
                              const float* __restrict__ Wr,
                              float* __restrict__ out,        // full d_out
                              float* __restrict__ probs_ws,
                              int*   __restrict__ eidx_ws)
{
    const int lane  = threadIdx.x & 31;
    const int token = blockIdx.x * 8 + (threadIdx.x >> 5);
    if (token >= kN) return;

    float acc[kE];
#pragma unroll
    for (int e = 0; e < kE; ++e) acc[e] = 0.f;

    const float* xr = x + (size_t)token * kD;
    for (int i = lane; i < kD; i += 32) {
        const float xv = xr[i];
        const float* wr = Wr + (size_t)i * kE;
#pragma unroll
        for (int e = 0; e < kE; ++e) acc[e] += xv * wr[e];
    }
#pragma unroll
    for (int off = 16; off > 0; off >>= 1) {
#pragma unroll
        for (int e = 0; e < kE; ++e) acc[e] += __shfl_xor(acc[e], off, 32);
    }

    if (lane == 0) {
        float mx = acc[0]; int best = 0;
#pragma unroll
        for (int e = 1; e < kE; ++e) { if (acc[e] > mx) { mx = acc[e]; best = e; } }
        float s = 0.f;
#pragma unroll
        for (int e = 0; e < kE; ++e) s += __expf(acc[e] - mx);
        const float maxp = 1.f / s;

        float* logits = out + (size_t)kN * kD;
#pragma unroll
        for (int e = 0; e < kE; ++e) logits[(size_t)token * kE + e] = acc[e];
        out[(size_t)kN * kD + (size_t)kN * kE + token] = (float)best;
        probs_ws[token] = maxp;
        eidx_ws[token]  = best;
    }
}

// ---------------------------------------------------------------------------
// Kernel 2: bucket tokens per expert.
// ---------------------------------------------------------------------------
__global__ void init_counts_kernel(int* __restrict__ counts)
{
    if (threadIdx.x < kE) counts[threadIdx.x] = 0;
}

__global__ void bucket_kernel(const int* __restrict__ eidx,
                              int* __restrict__ counts,
                              int* __restrict__ tlist)
{
    const int t = blockIdx.x * 256 + threadIdx.x;
    if (t < kN) {
        const int e   = eidx[t];
        const int pos = atomicAdd(&counts[e], 1);
        tlist[e * kN + pos] = t;
    }
}

// ---------------------------------------------------------------------------
// Kernel 3: fused routed FFN. grid = (N/32, E), 256 thr = 8 waves.
// TDM gathers 32 token rows (f32) into LDS; per F-chunk of 128:
//   stage A: h[32,128] = relu(X @ Wi[:,chunk])  (2 M-tiles/wave, K=768)
//   stage B: out[32,768] += h @ Wo[chunk,:]     (2 M-tiles x 6 N-tiles/wave)
// Every Wi/Wo B-fragment feeds 2 WMMAs (M reuse). Accumulators (12 x v8f)
// live in VGPRs across all 24 chunks.
// ---------------------------------------------------------------------------
__global__ __launch_bounds__(256, 2)
void moe_ffn_kernel(const float* __restrict__ x,
                    const float* __restrict__ Wi,
                    const float* __restrict__ Wo,
                    const int*   __restrict__ counts,
                    const int*   __restrict__ tlist,
                    const float* __restrict__ probs,
                    float* __restrict__ out)
{
    const int e    = blockIdx.y;
    const int tile = blockIdx.x;
    const int cnt  = counts[e];
    if (tile * kTileM >= cnt) return;

    const int tid  = threadIdx.x;
    const int wave = tid >> 5;
    const int lane = tid & 31;
    const int nl   = lane & 15;                  // N/M position in half-wave
    const int g    = lane >> 4;                  // half-wave select (ISA layout)

    // x_stage rows padded to 780 f32 by TDM (pad_interval=256dw, amount=4dw):
    // 16B-aligned rows, stride % 64 banks == 12 -> conflict-free b128 reads.
    __shared__ __align__(16) float  x_stage[kTileM][kXPitch];        // ~97.5 KB
    __shared__ __align__(16) __bf16 h_tile[kTileM][kFChunk + 8];     // ~8.5 KB
    __shared__ int   s_token[kTileM];
    __shared__ float s_prob[kTileM];

    if (tid < kTileM) {
        const int r   = tile * kTileM + tid;
        const int tok = (r < cnt) ? tlist[e * kN + r] : -1;
        s_token[tid] = tok;
        s_prob[tid]  = (tok >= 0) ? probs[tok] : 0.f;
    }
    __syncthreads();

    // ---- TDM gather of 32 scattered token rows into LDS (wave 0 issues; TDM
    // ignores EXEC so it must sit in a wave-uniform branch). Two descriptors,
    // 16 x 16-bit row indices each. Invalid rows clamp to row 0 (never stored).
    if (wave == 0) {
        int pk = 0;
        if (lane < 16) {
            int t0 = s_token[2 * lane];     if (t0 < 0) t0 = 0;
            int t1 = s_token[2 * lane + 1]; if (t1 < 0) t1 = 0;
            pk = (t0 & 0xffff) | (t1 << 16);
        }
        const unsigned lds0 = (unsigned)(size_t)(&x_stage[0][0]);
        const unsigned long long ga = (unsigned long long)(size_t)x;

        // group0: count=1, gather_mode=1 (16-bit idx), lds_addr, global_addr, type=2
        u32x4 g0a = { 1u | (1u << 31), lds0,
                      (unsigned)ga, (unsigned)(ga >> 32) | (2u << 30) };
        // group1: data_size=4B(2), pad_enable, pad_interval=256dw(7), pad_amount=4dw(3),
        //         tensor_dim0=768, tensor_dim1=N, tile_dim0=768, tile_dim1=16 idx,
        //         tensor_dim0_stride=768
        i32x8 g1 = { (int)((2u << 16) | (1u << 20) | (7u << 22) | (3u << 25)),
                     (int)(768u << 16),       // [63:48] tensor_dim0 lo16
                     (int)((unsigned)kN << 16), // [95:80] tensor_dim1 lo16
                     (int)(768u << 16),       // [127:112] tile_dim0
                     16,                      // [143:128] tile_dim1 = #indices
                     768, 0, 0 };             // tensor_dim0_stride = 768
        i32x4 g2a = { __builtin_amdgcn_readlane(pk, 0), __builtin_amdgcn_readlane(pk, 1),
                      __builtin_amdgcn_readlane(pk, 2), __builtin_amdgcn_readlane(pk, 3) };
        i32x4 g3a = { __builtin_amdgcn_readlane(pk, 4), __builtin_amdgcn_readlane(pk, 5),
                      __builtin_amdgcn_readlane(pk, 6), __builtin_amdgcn_readlane(pk, 7) };
        TDM_LOAD(g0a, g1, g2a, g3a);

        u32x4 g0b = g0a;
        g0b[1] = lds0 + 16u * kXPitch * 4u;   // rows 16..31
        i32x4 g2b = { __builtin_amdgcn_readlane(pk, 8),  __builtin_amdgcn_readlane(pk, 9),
                      __builtin_amdgcn_readlane(pk, 10), __builtin_amdgcn_readlane(pk, 11) };
        i32x4 g3b = { __builtin_amdgcn_readlane(pk, 12), __builtin_amdgcn_readlane(pk, 13),
                      __builtin_amdgcn_readlane(pk, 14), __builtin_amdgcn_readlane(pk, 15) };
        TDM_LOAD(g0b, g1, g2b, g3b);

        __builtin_amdgcn_s_wait_tensorcnt(0);
    }

    const float* wiE = Wi + (size_t)e * kD * kF;
    const float* woE = Wo + (size_t)e * kF * kD;

    v8f acc[12];                                  // [nt][mt] out accumulators
#pragma unroll
    for (int t = 0; t < 12; ++t) acc[t] = {0.f,0.f,0.f,0.f,0.f,0.f,0.f,0.f};

    for (int fc = 0; fc < kF; fc += kFChunk) {
        __syncthreads();   // TDM data visible (iter 0) / stage-B h reads done

        // -------- stage A: wave owns h columns [wave*16, wave*16+16) --------
        {
            v8f h0 = {0.f,0.f,0.f,0.f,0.f,0.f,0.f,0.f};
            v8f h1 = {0.f,0.f,0.f,0.f,0.f,0.f,0.f,0.f};
            const int colA = fc + wave * 16 + nl;
#pragma unroll 1
            for (int ks = 0; ks < kD; ks += 32) {
                // B (32x16 bf16): lane = column, K rows g*8+{0..7} and 16+g*8+{0..7}.
                ABFrag b;
                const float* src = wiE + (size_t)(ks + g * 8) * kF + colA;
#pragma unroll
                for (int j = 0; j < 16; ++j) b.v[j] = f2bf(src[(size_t)((j & 7) + (j >> 3) * 16) * kF]);
                // A: two 16-row tiles from the f32 x_stage, converted at use.
#pragma unroll
                for (int mt = 0; mt < 2; ++mt) {
                    const float* xr = &x_stage[mt * 16 + nl][ks + g * 8];
                    float4 q0 = *reinterpret_cast<const float4*>(xr);
                    float4 q1 = *reinterpret_cast<const float4*>(xr + 4);
                    float4 q2 = *reinterpret_cast<const float4*>(xr + 16);
                    float4 q3 = *reinterpret_cast<const float4*>(xr + 20);
                    ABFrag a;
                    a.v[0]=f2bf(q0.x); a.v[1]=f2bf(q0.y); a.v[2]=f2bf(q0.z); a.v[3]=f2bf(q0.w);
                    a.v[4]=f2bf(q1.x); a.v[5]=f2bf(q1.y); a.v[6]=f2bf(q1.z); a.v[7]=f2bf(q1.w);
                    a.v[8]=f2bf(q2.x); a.v[9]=f2bf(q2.y); a.v[10]=f2bf(q2.z); a.v[11]=f2bf(q2.w);
                    a.v[12]=f2bf(q3.x); a.v[13]=f2bf(q3.y); a.v[14]=f2bf(q3.z); a.v[15]=f2bf(q3.w);
                    if (mt == 0)
                        h0 = __builtin_amdgcn_wmma_f32_16x16x32_bf16(
                                 false, a.v, false, b.v, (short)0, h0, false, false);
                    else
                        h1 = __builtin_amdgcn_wmma_f32_16x16x32_bf16(
                                 false, a.v, false, b.v, (short)0, h1, false, false);
                }
            }
#pragma unroll
            for (int r = 0; r < 8; ++r) {          // D layout: (m=r+8g, n=nl)
                const float v0 = h0[r];
                const float v1 = h1[r];
                h_tile[r + 8 * g][wave * 16 + nl]      = f2bf(v0 > 0.f ? v0 : 0.f);
                h_tile[16 + r + 8 * g][wave * 16 + nl] = f2bf(v1 > 0.f ? v1 : 0.f);
            }
        }
        __syncthreads();

        // -------- stage B: wave owns out columns [wave*96, wave*96+96) ------
#pragma unroll 1
        for (int kf = 0; kf < kFChunk; kf += 32) {
            ABFrag a0, a1;
            a0.q[0] = *reinterpret_cast<const uint4*>(&h_tile[nl][kf + g * 8]);
            a0.q[1] = *reinterpret_cast<const uint4*>(&h_tile[nl][kf + 16 + g * 8]);
            a1.q[0] = *reinterpret_cast<const uint4*>(&h_tile[16 + nl][kf + g * 8]);
            a1.q[1] = *reinterpret_cast<const uint4*>(&h_tile[16 + nl][kf + 16 + g * 8]);
            const float* srcB = woE + (size_t)(fc + kf + g * 8) * kD + wave * 96 + nl;
#pragma unroll
            for (int nt = 0; nt < 6; ++nt) {
                ABFrag b;                          // fetched once, used by 2 WMMAs
#pragma unroll
                for (int j = 0; j < 16; ++j)
                    b.v[j] = f2bf(srcB[(size_t)((j & 7) + (j >> 3) * 16) * kD + nt * 16]);
                acc[nt * 2 + 0] = __builtin_amdgcn_wmma_f32_16x16x32_bf16(
                                      false, a0.v, false, b.v, (short)0,
                                      acc[nt * 2 + 0], false, false);
                acc[nt * 2 + 1] = __builtin_amdgcn_wmma_f32_16x16x32_bf16(
                                      false, a1.v, false, b.v, (short)0,
                                      acc[nt * 2 + 1], false, false);
            }
        }
    }

    // Epilogue: gate-scale and scatter; each token row written exactly once.
#pragma unroll
    for (int nt = 0; nt < 6; ++nt) {
        const int col = wave * 96 + nt * 16 + nl;
#pragma unroll
        for (int mt = 0; mt < 2; ++mt) {
#pragma unroll
            for (int r = 0; r < 8; ++r) {
                const int m   = mt * 16 + r + 8 * g;
                const int tok = s_token[m];
                if (tok >= 0)
                    out[(size_t)tok * kD + col] = acc[nt * 2 + mt][r] * s_prob[m];
            }
        }
    }
}

// ---------------------------------------------------------------------------
// Launcher. d_in: [hidden_states, Wr, Wi, Wo] (all f32).
// d_out: [N*D hidden_out | N*E router_logits | N expert_index] floats.
// ---------------------------------------------------------------------------
extern "C" void kernel_launch(void* const* d_in, const int* in_sizes, int n_in,
                              void* d_out, int out_size, void* d_ws, size_t ws_size,
                              hipStream_t stream)
{
    (void)in_sizes; (void)n_in; (void)out_size; (void)ws_size;

    const float* x  = (const float*)d_in[0];
    const float* Wr = (const float*)d_in[1];
    const float* Wi = (const float*)d_in[2];
    const float* Wo = (const float*)d_in[3];
    float* out = (float*)d_out;

    char* ws = (char*)d_ws;
    int*   counts   = (int*)ws;                                   // 8 ints
    int*   tlist    = (int*)(ws + 256);                           // E*N ints
    float* probs_ws = (float*)(ws + 256 + sizeof(int) * kE * kN); // N floats
    int*   eidx_ws  = (int*)(ws + 256 + sizeof(int) * kE * kN
                                 + sizeof(float) * kN);           // N ints

    init_counts_kernel<<<1, 32, 0, stream>>>(counts);
    router_kernel<<<kN / 8, 256, 0, stream>>>(x, Wr, out, probs_ws, eidx_ws);
    bucket_kernel<<<kN / 256, 256, 0, stream>>>(eidx_ws, counts, tlist);

    dim3 grid(kN / kTileM, kE);   // worst case: all tokens on one expert
    moe_ffn_kernel<<<grid, 256, 0, stream>>>(x, Wi, Wo, counts, tlist, probs_ws, out);
}
